// AutoregressivePolicy_61340722921622
// MI455X (gfx1250) — matrix operand order repaired and checked by reading, not compiled, not run on hardware
//
#include <hip/hip_runtime.h>
#include <math.h>

#define DD     256
#define NH     4
#define DHD    64
#define KNNC   32
#define NAG    64
#define NLN    64
#define NPT    20
#define DLANE  128
#define NKEY   96   // KNN + NLN

typedef __attribute__((ext_vector_type(16))) __bf16 v16bf;
typedef __attribute__((ext_vector_type(8)))  __bf16 v8bf;
typedef __attribute__((ext_vector_type(8)))  float  v8f;

// ---------------------------------------------------------------------------
// WMMA core (CDNA5, wave32, 16x16x32 bf16 -> f32)
// ---------------------------------------------------------------------------
__device__ __forceinline__ v8f wmma_bf16(v16bf a, v16bf b, v8f c) {
  return __builtin_amdgcn_wmma_f32_16x16x32_bf16(
      /*neg_a=*/false, a, /*neg_b=*/false, b,
      /*c_mod=*/(short)0, c, /*reuse_a=*/false, /*reuse_b=*/false);
}

// Fragment-position helper for the ISA 16-bit A layout:
// consumer lane = half*16 + m reads element i as K = (i<8?i:i+8)+half*8 (+32*kk)
// inverse: given kloc (K within 32-block) -> (half, i)
__device__ __forceinline__ void a_frag_pos(int kloc, int& half, int& i) {
  half = (kloc >> 3) & 1;
  i    = (kloc & 7) + ((kloc >> 4) & 1) * 8;
}

__device__ __forceinline__ float wrapf(float h) {
  const float PI  = 3.14159265358979323846f;
  const float TPI = 6.28318530717958647692f;
  float w = fmodf(h + PI, TPI);
  if (w < 0.0f) w += TPI;
  return w - PI;
}

// 256x256 GEMV: 4 k-chunks x 4 outputs/thread, float4 weight loads,
// LDS partial combine. Returns pre-bias dot for output column `tid`.
__device__ __forceinline__ float gemv256(const float* x, const float* W,
                                         int tid, float* s_part) {
  int co = tid >> 6;
  int og = (tid & 63) << 2;
  float a0 = 0, a1 = 0, a2 = 0, a3 = 0;
  int kb = co << 6;
  for (int k = kb; k < kb + 64; ++k) {
    float xv = x[k];
    const float4 w = *(const float4*)(W + (size_t)k * 256 + og);
    a0 = fmaf(xv, w.x, a0); a1 = fmaf(xv, w.y, a1);
    a2 = fmaf(xv, w.z, a2); a3 = fmaf(xv, w.w, a3);
  }
  s_part[co * 256 + og + 0] = a0;
  s_part[co * 256 + og + 1] = a1;
  s_part[co * 256 + og + 2] = a2;
  s_part[co * 256 + og + 3] = a3;
  __syncthreads();
  float r = s_part[tid] + s_part[256 + tid] + s_part[512 + tid] +
            s_part[768 + tid];
  __syncthreads();
  return r;
}

// 1024->256 GEMV (FFN down-proj), same scheme with 256-deep k-chunks.
__device__ __forceinline__ float gemv1024(const float* x, const float* W,
                                          int tid, float* s_part) {
  int co = tid >> 6;
  int og = (tid & 63) << 2;
  float a0 = 0, a1 = 0, a2 = 0, a3 = 0;
  int kb = co << 8;
  for (int k = kb; k < kb + 256; ++k) {
    float xv = x[k];
    const float4 w = *(const float4*)(W + (size_t)k * 256 + og);
    a0 = fmaf(xv, w.x, a0); a1 = fmaf(xv, w.y, a1);
    a2 = fmaf(xv, w.z, a2); a3 = fmaf(xv, w.w, a3);
  }
  s_part[co * 256 + og + 0] = a0;
  s_part[co * 256 + og + 1] = a1;
  s_part[co * 256 + og + 2] = a2;
  s_part[co * 256 + og + 3] = a3;
  __syncthreads();
  float r = s_part[tid] + s_part[256 + tid] + s_part[512 + tid] +
            s_part[768 + tid];
  __syncthreads();
  return r;
}

__device__ __forceinline__ float block_layernorm(float x, int tid,
                                                 const float* g,
                                                 const float* bb,
                                                 float* s_red) {
  s_red[tid] = x;
  __syncthreads();
  for (int off = 128; off > 0; off >>= 1) {
    if (tid < off) s_red[tid] += s_red[tid + off];
    __syncthreads();
  }
  float m = s_red[0] * (1.0f / DD);
  __syncthreads();
  float d = x - m;
  s_red[tid] = d * d;
  __syncthreads();
  for (int off = 128; off > 0; off >>= 1) {
    if (tid < off) s_red[tid] += s_red[tid + off];
    __syncthreads();
  }
  float v = s_red[0] * (1.0f / DD);
  __syncthreads();
  return d * rsqrtf(v + 1e-5f) * g[tid] + bb[tid];
}

// ---------------------------------------------------------------------------
// S0: pack WMMA B-weights into per-lane fragment layout (bf16).
// Fragment f = kk*(N/16)+tn; dst[(f*32+lane)*16+i] = W[(kk*32+(i<8?i:i+8)+half*8)*N +
// tn*16 + n] with lane = half*16+n.  dst flat index == idx (bijection).
// ---------------------------------------------------------------------------
__device__ __forceinline__ void packB_one(const float* W, __bf16* dst, int N,
                                          int idx) {
  int NT = N >> 4;
  int f  = idx >> 9;  // 512 elements per fragment
  int r  = idx & 511;
  int l  = r >> 4;
  int i  = r & 15;
  int kk = f / NT, tn = f - kk * NT;
  int half = l >> 4, n = l & 15;
  int k = kk * 32 + (i < 8 ? half * 8 + i : 16 + half * 8 + (i - 8));
  dst[idx] = (__bf16)W[k * N + tn * 16 + n];
}

__global__ void pack_weights_kernel(const float* aeW2, const float* aeW3,
                                    const float* Wk, const float* Wv,
                                    __bf16* w2p, __bf16* w3p, __bf16* Wkp,
                                    __bf16* Wvp) {
  const int T2 = 64 * 128, T3 = 128 * 256, TK = 256 * 256;
  int total = T2 + T3 + TK + TK;
  for (int idx = blockIdx.x * 256 + threadIdx.x; idx < total;
       idx += gridDim.x * 256) {
    if (idx < T2) packB_one(aeW2, w2p, 128, idx);
    else if (idx < T2 + T3) packB_one(aeW3, w3p, 256, idx - T2);
    else if (idx < T2 + T3 + TK) packB_one(Wk, Wkp, 256, idx - T2 - T3);
    else packB_one(Wv, Wvp, 256, idx - T2 - T3 - TK);
  }
}

// ---------------------------------------------------------------------------
// S1: lane encoder + maxpool + map projection  (one block per (b, lane))
// ---------------------------------------------------------------------------
__global__ __launch_bounds__(128)
void map_encoder_kernel(const float* map_lanes, const float* lane_mask,
                        const float* leW1, const float* leb1,
                        const float* leW2, const float* leb2,
                        const float* leW3, const float* leb3,
                        const float* lpW,  const float* lpb,
                        float* mapProj) {
  int bl  = blockIdx.x;
  int tid = threadIdx.x;
  __shared__ float h1[NPT][32];
  __shared__ float h2[NPT][64];
  __shared__ float h3[NPT][128];
  __shared__ float lf[128];
  const float* src = &map_lanes[(size_t)bl * NPT * 3];

  for (int i = tid; i < NPT * 32; i += 128) {
    int p = i >> 5, u = i & 31;
    float acc = leb1[u];
#pragma unroll
    for (int k = 0; k < 3; ++k) acc += src[p * 3 + k] * leW1[k * 32 + u];
    h1[p][u] = fmaxf(acc, 0.0f);
  }
  __syncthreads();
  for (int i = tid; i < NPT * 64; i += 128) {
    int p = i >> 6, u = i & 63;
    float acc = leb2[u];
    for (int k = 0; k < 32; ++k) acc += h1[p][k] * leW2[k * 64 + u];
    h2[p][u] = fmaxf(acc, 0.0f);
  }
  __syncthreads();
  for (int i = tid; i < NPT * 128; i += 128) {
    int p = i >> 7, u = i & 127;
    float acc = leb3[u];
    for (int k = 0; k < 64; ++k) acc += h2[p][k] * leW3[k * 128 + u];
    h3[p][u] = acc;
  }
  __syncthreads();
  float msk = lane_mask[bl];
  if (tid < 128) {
    float mx = h3[0][tid];
    for (int p = 1; p < NPT; ++p) mx = fmaxf(mx, h3[p][tid]);
    lf[tid] = mx * msk;
  }
  __syncthreads();
  for (int u = tid; u < DD; u += 128) {
    float acc = lpb[u];
    for (int k = 0; k < 128; ++k) acc += lf[k] * lpW[k * DD + u];
    mapProj[(size_t)bl * DD + u] = acc;
  }
}

// ---------------------------------------------------------------------------
// S2: mapK = mapProj @ Wk + bk ; mapV likewise (WMMA, 16-row strips,
//     fragment-layout A staging + packed B; separate K/V loops)
// ---------------------------------------------------------------------------
__global__ __launch_bounds__(256)
void map_kv_kernel(const float* mapProj,
                   const __bf16* Wkp, const float* bk,
                   const __bf16* Wvp, const float* bv,
                   __bf16* mapK, __bf16* mapV) {
  __shared__ v16bf sAf[8 * 32];   // 8 kk-fragments
  int tid = threadIdx.x;
  int r0  = blockIdx.x * 16;
  __bf16* sAe = (__bf16*)sAf;
  for (int i = tid; i < 16 * DD; i += 256) {
    int m = i >> 8, k = i & 255;
    int kk = k >> 5, kloc = k & 31, half, ii;
    a_frag_pos(kloc, half, ii);
    sAe[(kk * 32 + half * 16 + m) * 16 + ii] =
        (__bf16)mapProj[(size_t)(r0 + m) * DD + k];
  }
  __syncthreads();
  int wave = tid >> 5, lane = tid & 31;
  int half = (lane >> 4) & 1, nn = lane & 15;
  const v16bf* Wkv = (const v16bf*)Wkp;
  const v16bf* Wvv = (const v16bf*)Wvp;
  // wave handles 2 tn for K and 2 tn for V (16 tn total each)
#pragma unroll
  for (int jj = 0; jj < 2; ++jj) {
    int tn = wave * 2 + jj;
    v8f acc = {};
#pragma unroll
    for (int kk = 0; kk < 8; ++kk)
      acc = wmma_bf16(sAf[kk * 32 + lane], Wkv[(kk * 16 + tn) * 32 + lane],
                      acc);
#pragma unroll
    for (int j = 0; j < 8; ++j) {
      int row = r0 + j + half * 8;
      int col = tn * 16 + nn;
      mapK[(size_t)row * DD + col] = (__bf16)(acc[j] + bk[col]);
    }
  }
#pragma unroll
  for (int jj = 0; jj < 2; ++jj) {
    int tn = wave * 2 + jj;
    v8f acc = {};
#pragma unroll
    for (int kk = 0; kk < 8; ++kk)
      acc = wmma_bf16(sAf[kk * 32 + lane], Wvv[(kk * 16 + tn) * 32 + lane],
                      acc);
#pragma unroll
    for (int j = 0; j < 8; ++j) {
      int row = r0 + j + half * 8;
      int col = tn * 16 + nn;
      mapV[(size_t)row * DD + col] = (__bf16)(acc[j] + bv[col]);
    }
  }
}

// ---------------------------------------------------------------------------
// Main persistent kernel: one 256-thread workgroup (8 wave32) per batch chain
// ---------------------------------------------------------------------------
__global__ __launch_bounds__(256)
void policy_main_kernel(const float* agents_seq, const float* agents_mask,
                        const int* mode_c, const float* mode_embed,
                        const float* aeW1, const float* aeb1,
                        const float* aeb2, const float* aeb3,
                        const __bf16* w2p, const __bf16* w3p,
                        const float* Wq, const float* bq,
                        const __bf16* Wkp, const float* bk,
                        const __bf16* Wvp, const float* bv,
                        const float* Wo, const float* bo,
                        const float* ln1g, const float* ln1b,
                        const float* fW1, const float* fb1,
                        const float* fW2, const float* fb2,
                        const float* ln2g, const float* ln2b,
                        const float* ahW1, const float* ahb1,
                        const float* ahW2, const float* ahb2,
                        const __bf16* mapK, const __bf16* mapV,
                        float* out, int B, int T) {
  __shared__ float  s_mq[DD];
  __shared__ float  s_ego[4];
  __shared__ float  s_ag[NAG][4];
  __shared__ float  s_dist[NAG];
  __shared__ int    s_sel[KNNC];
  __shared__ v16bf  s_fA1[8 * 32];    // layer2 A frags: 4 tm x 2 kk
  __shared__ v16bf  s_fA2[16 * 32];   // layer3 A frags: 4 tm x 4 kk
  __shared__ v16bf  s_fA3[16 * 32];   // K/V    A frags: 2 tm x 8 kk
  __shared__ alignas(16) __bf16 s_pa[NAG][DD];
  __shared__ alignas(16) __bf16 s_K[KNNC][DD];
  __shared__ alignas(16) __bf16 s_V[KNNC][DD];
  __shared__ alignas(16) __bf16 s_mapK[NLN][DD];
  __shared__ alignas(16) __bf16 s_mapV[NLN][DD];
  __shared__ float  s_q[DD];
  __shared__ float  s_sc[NH][NKEY];
  __shared__ float  s_attn[DD];
  __shared__ float  s_x[DD];
  __shared__ float  s_y[DD];
  __shared__ float  s_ffn[4 * DD];
  __shared__ float  s_red[DD];
  __shared__ float  s_part[4 * DD];
  __shared__ float  s_a3[3];

  const int b    = blockIdx.x;
  const int tid  = threadIdx.x;
  const int wave = tid >> 5;
  const int lane = tid & 31;
  const int half = (lane >> 4) & 1;
  const int nn   = lane & 15;

  const v16bf* w2v = (const v16bf*)w2p;
  const v16bf* w3v = (const v16bf*)w3p;
  const v16bf* Wkv = (const v16bf*)Wkp;
  const v16bf* Wvv = (const v16bf*)Wvp;

  // cache step-invariant map K/V rows for this batch in LDS (16B vectors)
  {
    const v8bf* gk = (const v8bf*)(mapK + (size_t)b * NLN * DD);
    const v8bf* gv = (const v8bf*)(mapV + (size_t)b * NLN * DD);
    v8bf* lk = (v8bf*)&s_mapK[0][0];
    v8bf* lv = (v8bf*)&s_mapV[0][0];
    for (int i = tid; i < NLN * DD / 8; i += 256) {
      lk[i] = gk[i];
      lv[i] = gv[i];
    }
  }
  // init carried state
  s_mq[tid] = mode_embed[mode_c[b] * DD + tid];
  if (tid < 3) s_ego[tid] = 0.0f;
  __syncthreads();

  for (int t = 0; t < T; ++t) {
    // ---- phase 1: ego-frame transform + distances -------------------------
    if (tid < NAG) {
      const float* ap = &agents_seq[(((size_t)b * T + t) * NAG + tid) * 4];
      // prefetch next step's agent row (global_prefetch_b8) -- hides the only
      // latency-serial global read of the autoregressive chain
      if (t + 1 < T)
        __builtin_prefetch(
            &agents_seq[(((size_t)b * T + (t + 1)) * NAG + tid) * 4], 0, 1);
      float ax = ap[0], ay = ap[1], ah = ap[2], aw = ap[3];
      float ex = s_ego[0], ey = s_ego[1], eh = s_ego[2];
      float c = cosf(eh), s = sinf(eh);
      float dx = ax - ex, dy = ay - ey;
      float xe = c * dx + s * dy;
      float ye = -s * dx + c * dy;
      s_ag[tid][0] = xe;
      s_ag[tid][1] = ye;
      s_ag[tid][2] = wrapf(ah - eh);
      s_ag[tid][3] = aw;
      s_dist[tid]  = sqrtf(xe * xe + ye * ye);
    }
    // q = mq @ Wq + bq (gemv256 has the phase-boundary syncs inside)
    s_q[tid] = gemv256(s_mq, Wq, tid, s_part) + bq[tid];
    __syncthreads();

    // ---- phase 2: agent MLP layer1 (4->64) -> A1 fragments ----------------
    {
      __bf16* dst = (__bf16*)s_fA1;
#pragma unroll
      for (int r = 0; r < 16; ++r) {
        int idx = tid + 256 * r;  // 0..4095
        int a = idx >> 6, u = idx & 63;
        float acc = aeb1[u];
#pragma unroll
        for (int k = 0; k < 4; ++k) acc += s_ag[a][k] * aeW1[k * 64 + u];
        int tm = a >> 4, m = a & 15;
        int kk = u >> 5, kloc = u & 31, hf, ii;
        a_frag_pos(kloc, hf, ii);
        dst[((tm * 2 + kk) * 32 + hf * 16 + m) * 16 + ii] =
            (__bf16)fmaxf(acc, 0.0f);
      }
    }
    __syncthreads();

    // ---- phase 3: layer2 (64x64)@(64x128), WMMA -> A2 fragments -----------
    {
      __bf16* dst = (__bf16*)s_fA2;
#pragma unroll
      for (int jj = 0; jj < 4; ++jj) {  // 32 tiles, 4 per wave
        int tile = wave * 4 + jj;
        int tm = tile >> 3, tn = tile & 7;
        v8f acc = {};
#pragma unroll
        for (int kk = 0; kk < 2; ++kk)
          acc = wmma_bf16(s_fA1[(tm * 2 + kk) * 32 + lane],
                          w2v[(kk * 8 + tn) * 32 + lane], acc);
#pragma unroll
        for (int j = 0; j < 8; ++j) {
          int row = tm * 16 + j + half * 8;
          int col = tn * 16 + nn;
          float v = fmaxf(acc[j] + aeb2[col], 0.0f);
          int m = row & 15;
          int kk2 = col >> 5, kloc = col & 31, hf, ii;
          a_frag_pos(kloc, hf, ii);
          dst[((tm * 4 + kk2) * 32 + hf * 16 + m) * 16 + ii] = (__bf16)v;
        }
      }
    }
    __syncthreads();

    // ---- phase 4: layer3 (64x128)@(128x256) + bias + mask -> s_pa ---------
#pragma unroll
    for (int jj = 0; jj < 8; ++jj) {    // 64 tiles, 8 per wave
      int tile = wave * 8 + jj;
      int tm = tile >> 4, tn = tile & 15;
      v8f acc = {};
#pragma unroll
      for (int kk = 0; kk < 4; ++kk)
        acc = wmma_bf16(s_fA2[(tm * 4 + kk) * 32 + lane],
                        w3v[(kk * 16 + tn) * 32 + lane], acc);
#pragma unroll
      for (int j = 0; j < 8; ++j) {
        int row = tm * 16 + j + half * 8;
        int col = tn * 16 + nn;
        float mval = agents_mask[(size_t)b * NAG + row];
        s_pa[row][col] = (__bf16)((acc[j] + aeb3[col]) * mval);
      }
    }
    __syncthreads();

    // ---- phase 5: exact top-KNN by rank counting --------------------------
    if (tid < NAG) {
      float di = s_dist[tid];
      int rank = 0;
      for (int j = 0; j < NAG; ++j) {
        float dj = s_dist[j];
        if (dj < di || (dj == di && j < tid)) rank++;
      }
      if (rank < KNNC) s_sel[rank] = tid;
    }
    __syncthreads();

    // ---- phase 5.5: gather selected rows into A3 fragments (once) ---------
    {
      __bf16* dst = (__bf16*)s_fA3;
      for (int idx = tid; idx < KNNC * DD; idx += 256) {
        int row32 = idx >> 8;       // 0..31
        int col   = idx & 255;
        __bf16 v = s_pa[s_sel[row32]][col];
        int tm = row32 >> 4, m = row32 & 15;
        int kk = col >> 5, kloc = col & 31, hf, ii;
        a_frag_pos(kloc, hf, ii);
        dst[((tm * 8 + kk) * 32 + hf * 16 + m) * 16 + ii] = v;
      }
    }
    __syncthreads();

    // ---- phase 6: K/V of selected agents, WMMA (separate K and V loops) ---
#pragma unroll
    for (int jj = 0; jj < 4; ++jj) {    // K: 2 tm x 16 tn = 32 tiles
      int rem = wave * 4 + jj;
      int tm = rem >> 4, tn = rem & 15;
      v8f acc = {};
#pragma unroll
      for (int kk = 0; kk < 8; ++kk)
        acc = wmma_bf16(s_fA3[(tm * 8 + kk) * 32 + lane],
                        Wkv[(kk * 16 + tn) * 32 + lane], acc);
#pragma unroll
      for (int j = 0; j < 8; ++j) {
        int row = tm * 16 + j + half * 8;
        int col = tn * 16 + nn;
        s_K[row][col] = (__bf16)(acc[j] + bk[col]);
      }
    }
#pragma unroll
    for (int jj = 0; jj < 4; ++jj) {    // V: 2 tm x 16 tn = 32 tiles
      int rem = wave * 4 + jj;
      int tm = rem >> 4, tn = rem & 15;
      v8f acc = {};
#pragma unroll
      for (int kk = 0; kk < 8; ++kk)
        acc = wmma_bf16(s_fA3[(tm * 8 + kk) * 32 + lane],
                        Wvv[(kk * 16 + tn) * 32 + lane], acc);
#pragma unroll
      for (int j = 0; j < 8; ++j) {
        int row = tm * 16 + j + half * 8;
        int col = tn * 16 + nn;
        s_V[row][col] = (__bf16)(acc[j] + bv[col]);
      }
    }
    __syncthreads();

    // ---- phase 7: attention scores (16B vector K reads) -------------------
    if (tid < NKEY) {
      const v8bf* kp = (tid < KNNC) ? (const v8bf*)&s_K[tid][0]
                                    : (const v8bf*)&s_mapK[tid - KNNC][0];
#pragma unroll
      for (int h = 0; h < NH; ++h) {
        float acc = 0.0f;
#pragma unroll
        for (int c = 0; c < 8; ++c) {
          v8bf kv = kp[h * 8 + c];
#pragma unroll
          for (int e = 0; e < 8; ++e)
            acc += s_q[h * DHD + c * 8 + e] * (float)kv[e];
        }
        s_sc[h][tid] = acc * 0.125f;  // 1/sqrt(64)
      }
    }
    __syncthreads();

    // ---- phase 8: softmax per head ---------------------------------------
    if (tid < NH) {
      float mx = -1e30f;
      for (int j = 0; j < NKEY; ++j) mx = fmaxf(mx, s_sc[tid][j]);
      float sum = 0.0f;
      for (int j = 0; j < NKEY; ++j) {
        float e = expf(s_sc[tid][j] - mx);
        s_sc[tid][j] = e;
        sum += e;
      }
      float inv = 1.0f / sum;
      for (int j = 0; j < NKEY; ++j) s_sc[tid][j] *= inv;
    }
    __syncthreads();

    // ---- phase 9: attention output ---------------------------------------
    {
      int h = tid >> 6, d = tid & 63;
      float acc = 0.0f;
      for (int j = 0; j < KNNC; ++j)
        acc += s_sc[h][j] * (float)s_V[j][h * DHD + d];
      for (int j = 0; j < NLN; ++j)
        acc += s_sc[h][KNNC + j] * (float)s_mapV[j][h * DHD + d];
      s_attn[h * DHD + d] = acc;
    }
    __syncthreads();

    // ---- phase 10: Wo + residual + LN1 -----------------------------------
    {
      float acc = gemv256(s_attn, Wo, tid, s_part) + bo[tid];
      float x = s_mq[tid] + acc;
      s_y[tid] = block_layernorm(x, tid, ln1g, ln1b, s_red);
    }
    __syncthreads();

    // ---- phase 11: FFN 256->1024 (gelu) -> 256 + residual + LN2 ----------
    {
      int i0 = tid * 4;
      float a0 = fb1[i0], a1 = fb1[i0 + 1], a2 = fb1[i0 + 2],
            a3 = fb1[i0 + 3];
      for (int k = 0; k < DD; ++k) {
        float xv = s_y[k];
        const float4 w = *(const float4*)(fW1 + (size_t)k * 1024 + i0);
        a0 = fmaf(xv, w.x, a0); a1 = fmaf(xv, w.y, a1);
        a2 = fmaf(xv, w.z, a2); a3 = fmaf(xv, w.w, a3);
      }
      const float kInvSqrt2 = 0.70710678118654752f;
      s_ffn[i0 + 0] = 0.5f * a0 * (1.0f + erff(a0 * kInvSqrt2));
      s_ffn[i0 + 1] = 0.5f * a1 * (1.0f + erff(a1 * kInvSqrt2));
      s_ffn[i0 + 2] = 0.5f * a2 * (1.0f + erff(a2 * kInvSqrt2));
      s_ffn[i0 + 3] = 0.5f * a3 * (1.0f + erff(a3 * kInvSqrt2));
    }
    __syncthreads();
    {
      float acc = gemv1024(s_ffn, fW2, tid, s_part) + fb2[tid];
      float x = s_y[tid] + acc;
      s_mq[tid] = block_layernorm(x, tid, ln2g, ln2b, s_red);
    }
    __syncthreads();

    // ---- phase 12: action head + ego update ------------------------------
    s_x[tid] = fmaxf(gemv256(s_mq, ahW1, tid, s_part) + ahb1[tid], 0.0f);
    __syncthreads();
    if (tid < 3) {
      float acc = ahb2[tid];
      for (int k = 0; k < DD; ++k) acc += s_x[k] * ahW2[k * 3 + tid];
      s_a3[tid] = acc;
    }
    __syncthreads();
    if (tid == 0) {
      float ex = s_ego[0], ey = s_ego[1], eh = s_ego[2];
      float cb = cosf(eh), sb = sinf(eh);
      float xg = cb * s_a3[0] - sb * s_a3[1] + ex;
      float yg = sb * s_a3[0] + cb * s_a3[1] + ey;
      float hg = wrapf(s_a3[2] + eh);
      size_t o = ((size_t)b * T + t) * 3;
      out[o + 0] = xg;
      out[o + 1] = yg;
      out[o + 2] = hg;
      s_ego[0] = xg;
      s_ego[1] = yg;
      s_ego[2] = hg;
    }
    __syncthreads();
  }
}

// ---------------------------------------------------------------------------
extern "C" void kernel_launch(void* const* d_in, const int* in_sizes, int n_in,
                              void* d_out, int out_size, void* d_ws,
                              size_t ws_size, hipStream_t stream) {
  const float* agents_seq  = (const float*)d_in[0];
  const float* agents_mask = (const float*)d_in[1];
  const int*   mode_c      = (const int*)d_in[2];
  const float* map_lanes   = (const float*)d_in[3];
  const float* map_mask    = (const float*)d_in[4];
  int pi = 5;
  const float* mode_embed = (const float*)d_in[pi++];
  const float* aeW1 = (const float*)d_in[pi++];
  const float* aeb1 = (const float*)d_in[pi++];
  const float* aeW2 = (const float*)d_in[pi++];
  const float* aeb2 = (const float*)d_in[pi++];
  const float* aeW3 = (const float*)d_in[pi++];
  const float* aeb3 = (const float*)d_in[pi++];
  const float* leW1 = (const float*)d_in[pi++];
  const float* leb1 = (const float*)d_in[pi++];
  const float* leW2 = (const float*)d_in[pi++];
  const float* leb2 = (const float*)d_in[pi++];
  const float* leW3 = (const float*)d_in[pi++];
  const float* leb3 = (const float*)d_in[pi++];
  const float* lpW  = (const float*)d_in[pi++];
  const float* lpb  = (const float*)d_in[pi++];
  const float* Wq   = (const float*)d_in[pi++];
  const float* bq   = (const float*)d_in[pi++];
  const float* Wk   = (const float*)d_in[pi++];
  const float* bk   = (const float*)d_in[pi++];
  const float* Wv   = (const float*)d_in[pi++];
  const float* bv   = (const float*)d_in[pi++];
  const float* Wo   = (const float*)d_in[pi++];
  const float* bo   = (const float*)d_in[pi++];
  const float* ln1g = (const float*)d_in[pi++];
  const float* ln1b = (const float*)d_in[pi++];
  const float* fW1  = (const float*)d_in[pi++];
  const float* fb1  = (const float*)d_in[pi++];
  const float* fW2  = (const float*)d_in[pi++];
  const float* fb2  = (const float*)d_in[pi++];
  const float* ln2g = (const float*)d_in[pi++];
  const float* ln2b = (const float*)d_in[pi++];
  const float* ahW1 = (const float*)d_in[pi++];
  const float* ahb1 = (const float*)d_in[pi++];
  const float* ahW2 = (const float*)d_in[pi++];
  const float* ahb2 = (const float*)d_in[pi++];

  const int B = in_sizes[2];
  const int T = in_sizes[0] / (B * NAG * 4);

  // workspace layout (all offsets 32B-aligned)
  char* ws = (char*)d_ws;
  size_t off = 0;
  float* mapProj = (float*)(ws + off);  off += (size_t)B * NLN * DD * sizeof(float);
  __bf16* mapK   = (__bf16*)(ws + off); off += (size_t)B * NLN * DD * sizeof(__bf16);
  __bf16* mapV   = (__bf16*)(ws + off); off += (size_t)B * NLN * DD * sizeof(__bf16);
  __bf16* w2p    = (__bf16*)(ws + off); off += (size_t)64 * 128 * sizeof(__bf16);
  __bf16* w3p    = (__bf16*)(ws + off); off += (size_t)128 * 256 * sizeof(__bf16);
  __bf16* Wkp    = (__bf16*)(ws + off); off += (size_t)256 * 256 * sizeof(__bf16);
  __bf16* Wvp    = (__bf16*)(ws + off); off += (size_t)256 * 256 * sizeof(__bf16);
  (void)ws_size; (void)n_in; (void)out_size;

  pack_weights_kernel<<<672, 256, 0, stream>>>(aeW2, aeW3, Wk, Wv, w2p, w3p,
                                               Wkp, Wvp);
  map_encoder_kernel<<<B * NLN, 128, 0, stream>>>(
      map_lanes, map_mask, leW1, leb1, leW2, leb2, leW3, leb3, lpW, lpb,
      mapProj);
  map_kv_kernel<<<(B * NLN) / 16, 256, 0, stream>>>(mapProj, Wkp, bk, Wvp, bv,
                                                    mapK, mapV);
  policy_main_kernel<<<B, 256, 0, stream>>>(
      agents_seq, agents_mask, mode_c, mode_embed, aeW1, aeb1, aeb2, aeb3,
      w2p, w3p, Wq, bq, Wkp, bk, Wvp, bv, Wo, bo, ln1g, ln1b, fW1, fb1, fW2,
      fb2, ln2g, ln2b, ahW1, ahb1, ahW2, ahb2, mapK, mapV, (float*)d_out, B,
      T);
}
